// OpenCLIPViT_42872363549282
// MI455X (gfx1250) — matrix operand order repaired and checked by reading, not compiled, not run on hardware
//
#include <hip/hip_runtime.h>
#include <hip/hip_bf16.h>

typedef __attribute__((ext_vector_type(16))) _Float16 v16h;
typedef __attribute__((ext_vector_type(8)))  _Float16 v8h;
typedef __attribute__((ext_vector_type(8)))  float    v8f;

#define BB     64
#define SEQ    257
#define DIM    1024
#define NH     16
#define HDIM   64
#define NPAD   288                 // 9*32, >= 257 (WMMA K-dim padding)
#define MROWS  (BB * SEQ)          // 16448 = 32 * 514 (exact 32-row blocks)
#define MBLK   (MROWS / 32)        // 514

union AU { v16h v; v8h h[2]; };

#define WMMA_F16(A, B, C) \
  __builtin_amdgcn_wmma_f32_16x16x32_f16(false, (A), false, (B), (short)0, (C), false, false)

// ---------------------------------------------------------------- convert
__global__ void cvt_f32_f16(const float* __restrict__ in, _Float16* __restrict__ out, int n) {
  int i = blockIdx.x * blockDim.x + threadIdx.x;
  if (i < n) out[i] = (_Float16)in[i];
}

// ---------------------------------------------------------------- QKV GEMM
// qkv[row, e] = sum_d x[row,d] * Wi[e,d] + bi[e] ; scatter to Q/K (scaled) / V^T
// One wave computes a 32x64 output block (2 row-tiles x 4 col-tiles): B frags
// are reused across both row tiles (0.75 KB of L2 traffic per WMMA).
__global__ __launch_bounds__(32)
void qkv_gemm(const _Float16* __restrict__ xh, const _Float16* __restrict__ wh,
              const float* __restrict__ bias,
              _Float16* __restrict__ qb, _Float16* __restrict__ kb,
              _Float16* __restrict__ vt)
{
  const int lane = threadIdx.x & 31;
  const int m  = lane & 15;
  const int hi = lane >> 4;
  const int row0 = blockIdx.y * 32;
  const int ct0  = blockIdx.x * 4;            // four 16-wide column tiles

  v8f acc[2][4] = {};
  const _Float16* arow0 = xh + (size_t)(row0 + m) * DIM;
  const _Float16* arow1 = arow0 + (size_t)16 * DIM;
  for (int k0 = 0; k0 < DIM; k0 += 32) {
    AU a0, a1;
    a0.h[0] = *(const v8h*)(arow0 + k0 + hi * 16);
    a0.h[1] = *(const v8h*)(arow0 + k0 + hi * 16 + 8);
    a1.h[0] = *(const v8h*)(arow1 + k0 + hi * 16);
    a1.h[1] = *(const v8h*)(arow1 + k0 + hi * 16 + 8);
#pragma unroll
    for (int t = 0; t < 4; ++t) {
      const int col = (ct0 + t) * 16 + m;
      v16h bf = *(const v16h*)(wh + (size_t)col * DIM + k0 + hi * 16);
      acc[0][t] = WMMA_F16(a0.v, bf, acc[0][t]);
      acc[1][t] = WMMA_F16(a1.v, bf, acc[1][t]);
    }
  }
#pragma unroll
  for (int rt = 0; rt < 2; ++rt) {
#pragma unroll
    for (int t = 0; t < 4; ++t) {
      const int col  = (ct0 + t) * 16 + m;
      const float bv = bias[col];
      const int part = col >> 10;             // 0=q 1=k 2=v
      const int rem  = col & 1023;
      const int head = rem >> 6;
      const int hd   = rem & 63;
#pragma unroll
      for (int r = 0; r < 8; ++r) {
        const int row  = row0 + rt * 16 + r + hi * 8;
        const int bidx = row / SEQ;
        const int nseq = row - bidx * SEQ;
        const int bh   = bidx * NH + head;
        const float v  = acc[rt][t][r] + bv;
        if (part == 0)      qb[((size_t)bh * NPAD + nseq) * HDIM + hd] = (_Float16)(v * 0.125f);
        else if (part == 1) kb[((size_t)bh * NPAD + nseq) * HDIM + hd] = (_Float16)v;
        else                vt[((size_t)bh * HDIM + hd) * NPAD + nseq] = (_Float16)v;
      }
    }
  }
}

// ---------------------------------------------------------------- attention
__global__ __launch_bounds__(32)
void attn_kernel(const _Float16* __restrict__ qb, const _Float16* __restrict__ kb,
                 const _Float16* __restrict__ vt, const unsigned char* __restrict__ mask,
                 _Float16* __restrict__ ao)
{
  __shared__ alignas(64) float    s_sc[16][292];    // padded stride: conflict-free
  __shared__ alignas(64) _Float16 s_p[16][NPAD];    // stride 288 keeps 16B align

  const int lane = threadIdx.x & 31;
  const int m  = lane & 15;
  const int hi = lane >> 4;
  const int qt = blockIdx.x;                  // 0..16
  const int h  = blockIdx.y;
  const int b  = blockIdx.z;
  const int bh = b * NH + h;

  const _Float16* qbase = qb + ((size_t)bh * NPAD + qt * 16) * HDIM;
  const _Float16* kbase = kb + (size_t)bh * NPAD * HDIM;
  const _Float16* vbase = vt + (size_t)bh * HDIM * NPAD;

  // ---- phase 1: S = (Q/8) K^T for all key tiles, mask, to LDS
  for (int kt = 0; kt < NPAD / 16; ++kt) {
    v8f c = {};
#pragma unroll
    for (int k0 = 0; k0 < HDIM; k0 += 32) {
      AU a;
      const _Float16* ar = qbase + m * HDIM + k0;
      a.h[0] = *(const v8h*)(ar + hi * 16);
      a.h[1] = *(const v8h*)(ar + hi * 16 + 8);
      v16h bf = *(const v16h*)(kbase + (size_t)(kt * 16 + m) * HDIM + k0 + hi * 16);
      c = WMMA_F16(a.v, bf, c);
    }
    const int kcol = kt * 16 + m;
    const bool dead = (kcol >= SEQ) || (mask[b * SEQ + kcol] != 0);
#pragma unroll
    for (int r = 0; r < 8; ++r)
      s_sc[r + hi * 8][kcol] = dead ? -1e30f : c[r];
  }
  __syncthreads();

  // ---- phase 2: two-pass softmax, one row per lane (lanes 0..15)
  if (lane < 16) {
    float mx = -1e30f;
    for (int j = 0; j < NPAD; ++j) mx = fmaxf(mx, s_sc[m][j]);
    float sum = 0.f;
    for (int j = 0; j < NPAD; ++j) {
      float e = __expf(s_sc[m][j] - mx);
      s_sc[m][j] = e;
      sum += e;
    }
    const float inv = 1.f / sum;
    for (int j = 0; j < NPAD; ++j)
      s_p[m][j] = (_Float16)(s_sc[m][j] * inv);
  }
  __syncthreads();

  // ---- phase 3: O = P V   (16 x 64)
  v8f o[4] = {};
  for (int k0 = 0; k0 < NPAD; k0 += 32) {
    AU a;
    a.h[0] = *(const v8h*)(&s_p[m][k0 + hi * 16]);
    a.h[1] = *(const v8h*)(&s_p[m][k0 + hi * 16 + 8]);
#pragma unroll
    for (int t = 0; t < 4; ++t) {
      const int hd = t * 16 + m;
      v16h bf = *(const v16h*)(vbase + (size_t)hd * NPAD + k0 + hi * 16);
      o[t] = WMMA_F16(a.v, bf, o[t]);
    }
  }
#pragma unroll
  for (int t = 0; t < 4; ++t) {
    const int hd = t * 16 + m;
#pragma unroll
    for (int r = 0; r < 8; ++r) {
      const int nseq = qt * 16 + r + hi * 8;
      if (nseq < SEQ)
        ao[((size_t)b * SEQ + nseq) * DIM + h * HDIM + hd] = (_Float16)o[t][r];
    }
  }
}

// ---------------------------------------------------------------- out proj
// One wave computes a 32x64 block, B frags reused across the two row tiles.
__global__ __launch_bounds__(32)
void out_gemm(const _Float16* __restrict__ ah, const _Float16* __restrict__ wh,
              const float* __restrict__ bias, float* __restrict__ out)
{
  const int lane = threadIdx.x & 31;
  const int m  = lane & 15;
  const int hi = lane >> 4;
  const int row0 = blockIdx.y * 32;
  const int ct0  = blockIdx.x * 4;

  v8f acc[2][4] = {};
  const _Float16* arow0 = ah + (size_t)(row0 + m) * DIM;
  const _Float16* arow1 = arow0 + (size_t)16 * DIM;
  for (int k0 = 0; k0 < DIM; k0 += 32) {
    AU a0, a1;
    a0.h[0] = *(const v8h*)(arow0 + k0 + hi * 16);
    a0.h[1] = *(const v8h*)(arow0 + k0 + hi * 16 + 8);
    a1.h[0] = *(const v8h*)(arow1 + k0 + hi * 16);
    a1.h[1] = *(const v8h*)(arow1 + k0 + hi * 16 + 8);
#pragma unroll
    for (int t = 0; t < 4; ++t) {
      const int col = (ct0 + t) * 16 + m;
      v16h bf = *(const v16h*)(wh + (size_t)col * DIM + k0 + hi * 16);
      acc[0][t] = WMMA_F16(a0.v, bf, acc[0][t]);
      acc[1][t] = WMMA_F16(a1.v, bf, acc[1][t]);
    }
  }
#pragma unroll
  for (int rt = 0; rt < 2; ++rt) {
#pragma unroll
    for (int t = 0; t < 4; ++t) {
      const int col  = (ct0 + t) * 16 + m;
      const float bv = bias[col];
#pragma unroll
      for (int r = 0; r < 8; ++r)
        out[(size_t)(row0 + rt * 16 + r + hi * 8) * DIM + col] = acc[rt][t][r] + bv;
    }
  }
}

// ---------------------------------------------------------------- launch
extern "C" void kernel_launch(void* const* d_in, const int* in_sizes, int n_in,
                              void* d_out, int out_size, void* d_ws, size_t ws_size,
                              hipStream_t stream)
{
  const float*         x    = (const float*)d_in[0];
  const unsigned char* mask = (const unsigned char*)d_in[1];  // jax bool -> 1 byte
  const float*         wi   = (const float*)d_in[2];
  const float*         bi   = (const float*)d_in[3];
  const float*         wo   = (const float*)d_in[4];
  const float*         bo   = (const float*)d_in[5];
  float*               out  = (float*)d_out;

  char* ws = (char*)d_ws;
  size_t off = 0;
  auto take = [&](size_t bytes) {
    char* p = ws + off;
    off += (bytes + 255) & ~(size_t)255;
    return p;
  };
  const size_t qkv_bytes = (size_t)BB * NH * NPAD * HDIM * sizeof(_Float16);
  _Float16* xh  = (_Float16*)take((size_t)MROWS * DIM * sizeof(_Float16));
  _Float16* wih = (_Float16*)take((size_t)3 * DIM * DIM * sizeof(_Float16));
  _Float16* woh = (_Float16*)take((size_t)DIM * DIM * sizeof(_Float16));
  _Float16* qb  = (_Float16*)take(qkv_bytes);
  _Float16* kb  = (_Float16*)take(qkv_bytes);
  _Float16* vt  = (_Float16*)take(qkv_bytes);
  _Float16* ao  = (_Float16*)take((size_t)MROWS * DIM * sizeof(_Float16));

  // zero padded Q/K/V rows (capture-safe)
  hipMemsetAsync(qb, 0, qkv_bytes, stream);
  hipMemsetAsync(kb, 0, qkv_bytes, stream);
  hipMemsetAsync(vt, 0, qkv_bytes, stream);

  const int n1 = MROWS * DIM;
  const int n2 = 3 * DIM * DIM;
  const int n3 = DIM * DIM;
  cvt_f32_f16<<<(n1 + 255) / 256, 256, 0, stream>>>(x,  xh,  n1);
  cvt_f32_f16<<<(n2 + 255) / 256, 256, 0, stream>>>(wi, wih, n2);
  cvt_f32_f16<<<(n3 + 255) / 256, 256, 0, stream>>>(wo, woh, n3);

  qkv_gemm<<<dim3(3 * DIM / 64, MBLK), 32, 0, stream>>>(xh, wih, bi, qb, kb, vt);
  attn_kernel<<<dim3((SEQ + 15) / 16, NH, BB), 32, 0, stream>>>(qb, kb, vt, mask, ao);
  out_gemm<<<dim3(DIM / 64, MBLK), 32, 0, stream>>>(ao, woh, bo, out);
}
